// GaussianGenerator_11613591568915
// MI455X (gfx1250) — matrix operand order repaired and checked by reading, not compiled, not run on hardware
//
#include <hip/hip_runtime.h>
#include <hip/hip_bf16.h>

typedef __attribute__((ext_vector_type(16))) __bf16 v16bf;
typedef __attribute__((ext_vector_type(8)))  __bf16 v8bf;
typedef __attribute__((ext_vector_type(8)))  float  v8f;

#define V_N 100000
#define D_N 300
#define K_N 200
#define B_N 256
#define DP  320   // D padded to 10*32
#define KP  224   // K padded to 14*16 == 7*32
#define OUTW (K_N + V_N)
#define VB  80    // v rows per transpose block (100000 = 80*1250)

static __device__ __forceinline__ v8f wmma_bf16(v16bf a, v16bf b, v8f c) {
  return __builtin_amdgcn_wmma_f32_16x16x32_bf16(false, a, false, b, (short)0, c, false, false);
}

// ---------------- Kernel 0: per-topic coefficients (tiny) ----------------
__global__ void topic_prep(const float* __restrict__ mu, const float* __restrict__ w,
                           const float* __restrict__ cd,
                           __bf16* __restrict__ Wq, __bf16* __restrict__ Wm,
                           __bf16* __restrict__ Wu,
                           float* __restrict__ Ck, float* __restrict__ capinv,
                           float* __restrict__ ctv) {
  int k = blockIdx.x * blockDim.x + threadIdx.x;
  if (k >= KP) return;
  if (k < K_N) {
    const float* murow = mu + k * D_N;
    const float* wrow  = w  + k * D_N;
    const float* cdrow = cd + k * D_N;
    float cap = 1.0f, ld = 0.0f, cmu = 0.0f, ct = 0.0f;
    for (int d = 0; d < D_N; ++d) {
      float cdv = cdrow[d];
      float di  = 1.0f / cdv;
      float wv  = wrow[d], m = murow[d];
      float u   = wv * di;
      cap += wv * u;
      ld  += logf(cdv);
      cmu += m * m * di;
      ct  += m * u;
      Wq[k * DP + d] = (__bf16)di;
      Wm[k * DP + d] = (__bf16)(-2.0f * m * di);
      Wu[k * DP + d] = (__bf16)u;
    }
    for (int d = D_N; d < DP; ++d) {
      Wq[k*DP+d] = (__bf16)0.f; Wm[k*DP+d] = (__bf16)0.f; Wu[k*DP+d] = (__bf16)0.f;
    }
    const float log2pi = 1.8378770664093453f;
    Ck[k]     = (float)D_N * log2pi + ld + logf(cap) + cmu;
    capinv[k] = 1.0f / cap;
    ctv[k]    = ct;
  } else {
    for (int d = 0; d < DP; ++d) {
      Wq[k*DP+d] = (__bf16)0.f; Wm[k*DP+d] = (__bf16)0.f; Wu[k*DP+d] = (__bf16)0.f;
    }
    Ck[k] = 0.f; capinv[k] = 0.f; ctv[k] = 0.f;
  }
}

// -------- Kernel 0b: sampled_x -> bf16, K padded to 224 (tiny) -----------
__global__ void sx_prep(const float* __restrict__ sx, __bf16* __restrict__ SXb) {
  int idx = blockIdx.x * 256 + threadIdx.x;
  if (idx >= B_N * KP) return;
  int b = idx / KP, k = idx % KP;
  SXb[idx] = (k < K_N) ? (__bf16)sx[b * K_N + k] : (__bf16)0.0f;
}

// ---- Kernel 0c (big-ws path): word_vectors -> padded bf16 X and X^2 -----
__global__ void x_prep(const float* __restrict__ wv,
                       __bf16* __restrict__ Xb, __bf16* __restrict__ X2b) {
  size_t idx = (size_t)blockIdx.x * 256 + threadIdx.x;   // V_N*DP = 32e6
  if (idx >= (size_t)V_N * DP) return;
  int v = (int)(idx / DP), d = (int)(idx % DP);
  float x = (d < D_N) ? wv[(size_t)v * D_N + d] : 0.0f;
  Xb[idx]  = (__bf16)x;
  X2b[idx] = (__bf16)(x * x);
}

// -- Kernel 1 (big-ws): logp; one wave = 4 v-tiles x 2 k-tiles (24 wmma/step)
__global__ __launch_bounds__(32) void logp_wmma4(
    const __bf16* __restrict__ Xb, const __bf16* __restrict__ X2b,
    const __bf16* __restrict__ Wq, const __bf16* __restrict__ Wm,
    const __bf16* __restrict__ Wu,
    const float* __restrict__ Ck, const float* __restrict__ capinv,
    const float* __restrict__ ctv,
    float* __restrict__ Lp) {
  const int kg = blockIdx.x;                 // 0..6 : k tiles 2kg, 2kg+1
  const int m0 = blockIdx.y * 64;            // 4 v-tiles
  const int lane = threadIdx.x & 31;
  const int lo = lane & 15, hi = lane >> 4;
  const int aoff = hi * 8;
  const int kbase = kg * 32 + lo;
  const __bf16* bq0 = Wq + (size_t)kbase * DP + hi * 8;
  const __bf16* bm0 = Wm + (size_t)kbase * DP + hi * 8;
  const __bf16* bu0 = Wu + (size_t)kbase * DP + hi * 8;

  int offs[4];                                // shared row offset for Xb and X2b
  #pragma unroll
  for (int vt = 0; vt < 4; ++vt) {
    int r = m0 + vt * 16 + lo;
    if (r > V_N - 1) r = V_N - 1;            // clamp (stores guarded below)
    offs[vt] = r * DP + aoff;
  }

  v8f acc12[4][2], acc3[4][2];
  v8f zero = {};
  #pragma unroll
  for (int vt = 0; vt < 4; ++vt)
    #pragma unroll
    for (int t = 0; t < 2; ++t) { acc12[vt][t] = zero; acc3[vt][t] = zero; }

  for (int d0 = 0; d0 < DP; d0 += 32) {
    v16bf av[4], a2v[4];
    #pragma unroll
    for (int vt = 0; vt < 4; ++vt) {
      v8bf x0 = *(const v8bf*)(Xb  + offs[vt] + d0);
      v8bf x1 = *(const v8bf*)(Xb  + offs[vt] + d0 + 16);
      v8bf y0 = *(const v8bf*)(X2b + offs[vt] + d0);
      v8bf y1 = *(const v8bf*)(X2b + offs[vt] + d0 + 16);
      #pragma unroll
      for (int e = 0; e < 8; ++e) {
        av[vt][e] = x0[e];  av[vt][8 + e] = x1[e];
        a2v[vt][e] = y0[e]; a2v[vt][8 + e] = y1[e];
      }
    }
    #pragma unroll
    for (int t = 0; t < 2; ++t) {
      const int toff = t * (16 * DP) + d0;
      v16bf b1 = __builtin_shufflevector(*(const v8bf*)(bq0 + toff), *(const v8bf*)(bq0 + toff + 16), 0,1,2,3,4,5,6,7,8,9,10,11,12,13,14,15);
      v16bf b2 = __builtin_shufflevector(*(const v8bf*)(bm0 + toff), *(const v8bf*)(bm0 + toff + 16), 0,1,2,3,4,5,6,7,8,9,10,11,12,13,14,15);
      v16bf b3 = __builtin_shufflevector(*(const v8bf*)(bu0 + toff), *(const v8bf*)(bu0 + toff + 16), 0,1,2,3,4,5,6,7,8,9,10,11,12,13,14,15);
      #pragma unroll
      for (int vt = 0; vt < 4; ++vt) {
        acc12[vt][t] = wmma_bf16(a2v[vt], b1, acc12[vt][t]);
        acc12[vt][t] = wmma_bf16(av[vt],  b2, acc12[vt][t]);
        acc3[vt][t]  = wmma_bf16(av[vt],  b3, acc3[vt][t]);
      }
    }
  }

  #pragma unroll
  for (int vt = 0; vt < 4; ++vt) {
    int vbase = m0 + vt * 16;
    if (vbase >= V_N) continue;              // uniform per tile
    #pragma unroll
    for (int t = 0; t < 2; ++t) {
      int bk = kbase + t * 16;
      if (bk < K_N) {
        float C = Ck[bk], ci = capinv[bk], ct = ctv[bk];
        float o[8];
        #pragma unroll
        for (int r = 0; r < 8; ++r) {
          float tt = acc3[vt][t][r] - ct;
          o[r] = -0.5f * (C + acc12[vt][t][r] - tt * tt * ci);
        }
        float4* p = (float4*)(Lp + (size_t)bk * V_N + vbase + hi * 8);
        p[0] = make_float4(o[0], o[1], o[2], o[3]);
        p[1] = make_float4(o[4], o[5], o[6], o[7]);
      }
    }
  }
}

// -- Kernel 1 (fallback, small ws): on-the-fly cvt, 1 v-tile x 7 k-tiles --
__global__ __launch_bounds__(32) void logp_wmma_fly(
    const float* __restrict__ wv,
    const __bf16* __restrict__ Wq, const __bf16* __restrict__ Wm,
    const __bf16* __restrict__ Wu,
    const float* __restrict__ Ck, const float* __restrict__ capinv,
    const float* __restrict__ ctv,
    float* __restrict__ Lp) {
  const int kg = blockIdx.x;
  const int m0 = blockIdx.y * 16;
  const int lane = threadIdx.x & 31;
  const int lo = lane & 15, hi = lane >> 4;
  const float* rowp = wv + (size_t)(m0 + lo) * D_N;
  const int aoff = hi * 8;
  const int kbase = kg * 112 + lo;
  const size_t tstep = (size_t)16 * DP;
  const __bf16* bq0 = Wq + (size_t)kbase * DP + hi * 8;
  const __bf16* bm0 = Wm + (size_t)kbase * DP + hi * 8;
  const __bf16* bu0 = Wu + (size_t)kbase * DP + hi * 8;

  v8f acc12[7], acc3[7];
  v8f zero = {};
  #pragma unroll
  for (int t = 0; t < 7; ++t) { acc12[t] = zero; acc3[t] = zero; }

  auto kstep = [&](v16bf av, v16bf a2v, int d0) {
    #pragma unroll
    for (int t = 0; t < 7; ++t) {
      const size_t toff = (size_t)t * tstep + d0;
      v16bf b1 = __builtin_shufflevector(*(const v8bf*)(bq0 + toff), *(const v8bf*)(bq0 + toff + 16), 0,1,2,3,4,5,6,7,8,9,10,11,12,13,14,15);
      v16bf b2 = __builtin_shufflevector(*(const v8bf*)(bm0 + toff), *(const v8bf*)(bm0 + toff + 16), 0,1,2,3,4,5,6,7,8,9,10,11,12,13,14,15);
      v16bf b3 = __builtin_shufflevector(*(const v8bf*)(bu0 + toff), *(const v8bf*)(bu0 + toff + 16), 0,1,2,3,4,5,6,7,8,9,10,11,12,13,14,15);
      acc12[t] = wmma_bf16(a2v, b1, acc12[t]);
      acc12[t] = wmma_bf16(av,  b2, acc12[t]);
      acc3[t]  = wmma_bf16(av,  b3, acc3[t]);
    }
  };

  for (int d0 = 0; d0 < 288; d0 += 32) {
    const float4* p1 = (const float4*)(rowp + d0 + aoff);
    const float4* p2 = (const float4*)(rowp + d0 + 16 + aoff);
    float4 x0 = p1[0], x1 = p1[1], y0 = p2[0], y1 = p2[1];
    float a[16] = { x0.x,x0.y,x0.z,x0.w, x1.x,x1.y,x1.z,x1.w,
                    y0.x,y0.y,y0.z,y0.w, y1.x,y1.y,y1.z,y1.w };
    v16bf av, a2v;
    #pragma unroll
    for (int e = 0; e < 16; ++e) { av[e] = (__bf16)a[e]; a2v[e] = (__bf16)(a[e]*a[e]); }
    kstep(av, a2v, d0);
  }
  {
    float a[16];
    #pragma unroll
    for (int e = 0; e < 8; ++e) {
      int c1 = 288 + aoff + e;
      int c2 = 288 + 16 + aoff + e;
      float x1v = rowp[c1 < D_N ? c1 : 0];
      float x2v = rowp[c2 < D_N ? c2 : 0];
      a[e]     = (c1 < D_N) ? x1v : 0.0f;
      a[8 + e] = (c2 < D_N) ? x2v : 0.0f;
    }
    v16bf av, a2v;
    #pragma unroll
    for (int e = 0; e < 16; ++e) { av[e] = (__bf16)a[e]; a2v[e] = (__bf16)(a[e]*a[e]); }
    kstep(av, a2v, 288);
  }

  #pragma unroll
  for (int t = 0; t < 7; ++t) {
    int bk = kbase + t * 16;
    if (bk < K_N) {
      float C = Ck[bk], ci = capinv[bk], ct = ctv[bk];
      float o[8];
      #pragma unroll
      for (int r = 0; r < 8; ++r) {
        float tt = acc3[t][r] - ct;
        o[r] = -0.5f * (C + acc12[t][r] - tt * tt * ci);
      }
      float4* p = (float4*)(Lp + (size_t)bk * V_N + m0 + hi * 8);
      p[0] = make_float4(o[0], o[1], o[2], o[3]);
      p[1] = make_float4(o[4], o[5], o[6], o[7]);
    }
  }
}

// ------------- Kernel 2: per-topic softmax max + inv(sum) ----------------
__global__ void softmax_stats(const float* __restrict__ Lp,
                              float* __restrict__ maxk, float* __restrict__ invsum) {
  int k = blockIdx.x;
  int t = threadIdx.x;
  __shared__ float sm[256];
  const float* row = Lp + (size_t)k * V_N;
  float m = -3.4e38f;
  for (int v = t; v < V_N; v += 256) m = fmaxf(m, row[v]);
  sm[t] = m; __syncthreads();
  for (int s = 128; s > 0; s >>= 1) { if (t < s) sm[t] = fmaxf(sm[t], sm[t + s]); __syncthreads(); }
  float mx = sm[0]; __syncthreads();
  float sum = 0.0f;
  for (int v = t; v < V_N; v += 256) sum += expf(row[v] - mx);
  sm[t] = sum; __syncthreads();
  for (int s = 128; s > 0; s >>= 1) { if (t < s) sm[t] += sm[t + s]; __syncthreads(); }
  if (t == 0) { maxk[k] = mx; invsum[k] = 1.0f / sm[0]; }
}

// - Kernel 2b: normalized probs -> v-major bf16, LDS-tiled transpose ------
__global__ __launch_bounds__(256) void probs_bf16(
    const float* __restrict__ Lp,
    const float* __restrict__ maxk, const float* __restrict__ invsum,
    __bf16* __restrict__ Pb) {
  __shared__ __bf16 sm[VB * KP];             // 35840 B
  const int v0 = blockIdx.x * VB;
  const int t = threadIdx.x;
  for (int idx = t; idx < KP * VB; idx += 256) {
    int k = idx / VB, vi = idx % VB;
    float p = 0.0f;
    if (k < K_N) p = expf(Lp[(size_t)k * V_N + v0 + vi] - maxk[k]) * invsum[k];
    sm[vi * KP + k] = (__bf16)p;
  }
  __syncthreads();
  for (int idx = t; idx < VB * KP; idx += 256) {
    Pb[(size_t)v0 * KP + idx] = sm[idx];
  }
}

// -- Kernel 3: word_dist via bf16 WMMA; one wave = 16 v-rows x 16 b-tiles --
__global__ __launch_bounds__(32) void out_wmma(
    const __bf16* __restrict__ Pb, const __bf16* __restrict__ SXb,
    float* __restrict__ out) {
  const int m0 = blockIdx.x * 16;
  const int lane = threadIdx.x & 31;
  const int lo = lane & 15, hi = lane >> 4;
  const __bf16* arow = Pb + (size_t)(m0 + lo) * KP;
  const int aoff = hi * 8;
  const __bf16* brow = SXb + (size_t)lo * KP + hi * 8;

  v8f acc[16];
  v8f zero = {};
  #pragma unroll
  for (int n = 0; n < 16; ++n) acc[n] = zero;

  for (int k0 = 0; k0 < KP; k0 += 32) {
    v8bf a0 = *(const v8bf*)(arow + k0 + aoff);
    v8bf a1 = *(const v8bf*)(arow + k0 + 16 + aoff);
    v16bf av;
    #pragma unroll
    for (int e = 0; e < 8; ++e) { av[e] = a0[e]; av[8 + e] = a1[e]; }
    #pragma unroll
    for (int n = 0; n < 16; ++n) {
      v16bf bv = __builtin_shufflevector(*(const v8bf*)(brow + (size_t)n * 16 * KP + k0), *(const v8bf*)(brow + (size_t)n * 16 * KP + k0 + 16), 0,1,2,3,4,5,6,7,8,9,10,11,12,13,14,15);
      acc[n] = wmma_bf16(av, bv, acc[n]);
    }
  }
  #pragma unroll
  for (int n = 0; n < 16; ++n) {
    float* op = out + (size_t)(n * 16 + lo) * OUTW + K_N + m0 + hi * 8;
    ((float4*)op)[0] = make_float4(acc[n][0], acc[n][1], acc[n][2], acc[n][3]);
    ((float4*)op)[1] = make_float4(acc[n][4], acc[n][5], acc[n][6], acc[n][7]);
  }
}

// ---------------- Kernel 4: concat prefix out[:, :K] = sx ----------------
__global__ void copy_sx(const float* __restrict__ sx, float* __restrict__ out) {
  int idx = blockIdx.x * 256 + threadIdx.x;
  if (idx >= B_N * K_N) return;
  int b = idx / K_N, k = idx % K_N;
  out[(size_t)b * OUTW + k] = sx[idx];
}

extern "C" void kernel_launch(void* const* d_in, const int* in_sizes, int n_in,
                              void* d_out, int out_size, void* d_ws, size_t ws_size,
                              hipStream_t stream) {
  const float* sx = (const float*)d_in[0];
  const float* wv = (const float*)d_in[1];
  const float* mu = (const float*)d_in[2];
  const float* cf = (const float*)d_in[3];
  const float* cd = (const float*)d_in[4];
  float* out = (float*)d_out;

  const size_t szLp  = (size_t)K_N * V_N * 4;   // 80.0 MB
  const size_t szX   = (size_t)V_N * DP * 2;    // 64.0 MB
  const size_t szPb  = (size_t)V_N * KP * 2;    // 44.8 MB
  const size_t szW   = (size_t)KP * DP * 2;
  const size_t szSX  = (size_t)B_N * KP * 2;
  const size_t smalls = 3 * szW + szSX + 5 * 1024;
  const size_t NEED_BIG = szLp + 2 * szX + smalls;   // X2b region also holds Pb

  char* ws = (char*)d_ws;
  bool big = (ws_size >= NEED_BIG);

  size_t off = 0;
  float*  Lp  = (float*)(ws + off);  off += szLp;
  __bf16* Xb  = nullptr;
  __bf16* Pb;                                   // Pb aliases X2b in big path
  __bf16* X2b = nullptr;
  if (big) {
    Xb  = (__bf16*)(ws + off); off += szX;
    X2b = (__bf16*)(ws + off); off += szX;      // max(szX, szPb) == szX
    Pb  = X2b;                                  // X2b dead before probs_bf16
  } else {
    Pb  = (__bf16*)(ws + off); off += szPb;
  }
  __bf16* Wq  = (__bf16*)(ws + off); off += szW;
  __bf16* Wm  = (__bf16*)(ws + off); off += szW;
  __bf16* Wu  = (__bf16*)(ws + off); off += szW;
  __bf16* SXb = (__bf16*)(ws + off); off += szSX;
  float* Ck     = (float*)(ws + off); off += 1024;
  float* capinv = (float*)(ws + off); off += 1024;
  float* ctv    = (float*)(ws + off); off += 1024;
  float* maxk   = (float*)(ws + off); off += 1024;
  float* invsum = (float*)(ws + off); off += 1024;

  topic_prep<<<1, 256, 0, stream>>>(mu, cf, cd, Wq, Wm, Wu, Ck, capinv, ctv);
  sx_prep<<<(B_N * KP + 255) / 256, 256, 0, stream>>>(sx, SXb);

  if (big) {
    x_prep<<<(int)(((size_t)V_N * DP + 255) / 256), 256, 0, stream>>>(wv, Xb, X2b);
    dim3 g1(7, (V_N / 16 + 3) / 4);   // k-group fast dim -> X tiles reused in L2
    logp_wmma4<<<g1, 32, 0, stream>>>(Xb, X2b, Wq, Wm, Wu, Ck, capinv, ctv, Lp);
  } else {
    dim3 g1(2, V_N / 16);
    logp_wmma_fly<<<g1, 32, 0, stream>>>(wv, Wq, Wm, Wu, Ck, capinv, ctv, Lp);
  }

  softmax_stats<<<K_N, 256, 0, stream>>>(Lp, maxk, invsum);
  probs_bf16<<<V_N / VB, 256, 0, stream>>>(Lp, maxk, invsum, Pb);

  out_wmma<<<V_N / 16, 32, 0, stream>>>(Pb, SXb, out);
  copy_sx<<<(B_N * K_N + 255) / 256, 256, 0, stream>>>(sx, out);
}
